// SpikingLinear2_66898410602865
// MI455X (gfx1250) — matrix operand order repaired and checked by reading, compiled, not run on hardware
//
#include <hip/hip_runtime.h>

typedef _Float16 v16h __attribute__((ext_vector_type(16)));
typedef _Float16 v8h  __attribute__((ext_vector_type(8)));
typedef _Float16 v4h  __attribute__((ext_vector_type(4)));
typedef float    v8f  __attribute__((ext_vector_type(8)));

#define T_STEPS 128
#define BATCH   256
#define D1      2048
#define D2      2048
#define TILE_M  64
#define TILE_N  64
#define KC      128                      // K-chunk (halves) staged per buffer
#define NCHUNK  (D1 / KC)                // 16 (even -> clean 2x software pipeline)
#define W_PITCH 2056                     // halves per W row in LDS (2048 + 8 pad, 16B-aligned rows)
#define X_PITCH 136                      // halves per x row in LDS (128 + 8 pad, 16B-aligned rows)
#define W_BYTES (TILE_N * W_PITCH * 2)   // 263168 B (resident across all T)
#define X_BYTES (TILE_M * X_PITCH * 2)   // 17408 B per buffer
#define SMEM_BYTES (W_BYTES + 2 * X_BYTES)  // 297984 B < 320 KB WGP LDS

__device__ __forceinline__ v16h cat8(v8h lo, v8h hi) {
    return __builtin_shufflevector(lo, hi, 0,1,2,3,4,5,6,7,8,9,10,11,12,13,14,15);
}

struct XRegs { float4 f[8]; };

// Phase 1: issue all 8 global_load_b128 as one clause (one base + i*64KB offsets).
// No wait here -- the caller runs a full WMMA chunk before consuming the data.
__device__ __forceinline__ void stage_load(const float* __restrict__ gsrc, XRegs& r) {
    #pragma unroll
    for (int i = 0; i < 8; ++i) {
        r.f[i] = *(const float4*)(gsrc + (size_t)i * 8 * D1);
        __builtin_prefetch(gsrc + (size_t)i * 8 * D1 + 2 * KC, 0, 0);  // chunk kc+2
    }
}

// Phase 2: f32 -> f16 convert + ds_store_b64 (one LDS base + i*2176 offsets).
__device__ __forceinline__ void stage_store(const XRegs& r, _Float16* __restrict__ ldst) {
    #pragma unroll
    for (int i = 0; i < 8; ++i) {
        v4h h = { (_Float16)r.f[i].x, (_Float16)r.f[i].y,
                  (_Float16)r.f[i].z, (_Float16)r.f[i].w };
        *(v4h*)(ldst + i * 8 * X_PITCH) = h;
    }
}

// 4 k-steps of v_wmma_f32_16x16x32_f16 over one staged chunk (2 accumulators).
__device__ __forceinline__ void compute_chunk(const _Float16* __restrict__ bRowPtr, int kcOff,
                                              const _Float16* __restrict__ aRow0Ptr,
                                              const _Float16* __restrict__ aRow1Ptr,
                                              int aOff, int bOff, v8f& cur0, v8f& cur1) {
    #pragma unroll
    for (int ks = 0; ks < KC / 32; ++ks) {
        const int kA = ks * 32 + aOff;
        const int kB = kcOff + ks * 32 + bOff;
        v16h bf = cat8(*(const v8h*)(bRowPtr  + kB), *(const v8h*)(bRowPtr  + kB + 16));
        v16h a0 = cat8(*(const v8h*)(aRow0Ptr + kA), *(const v8h*)(aRow0Ptr + kA + 16));
        v16h a1 = cat8(*(const v8h*)(aRow1Ptr + kA), *(const v8h*)(aRow1Ptr + kA + 16));
        cur0 = __builtin_amdgcn_wmma_f32_16x16x32_f16(false, a0, false, bf,
                                                      (short)0, cur0, false, false);
        cur1 = __builtin_amdgcn_wmma_f32_16x16x32_f16(false, a1, false, bf,
                                                      (short)0, cur1, false, false);
    }
}

__global__ __launch_bounds__(256, 1)
void snn_spiking_linear_kernel(const float* __restrict__ x,
                               const float* __restrict__ W,
                               float* __restrict__ out) {
    extern __shared__ char smem[];
    _Float16* sW  = (_Float16*)smem;                          // [TILE_N][W_PITCH], persistent
    _Float16* sX0 = (_Float16*)(smem + W_BYTES);              // x chunk buffer 0
    _Float16* sX1 = (_Float16*)(smem + W_BYTES + X_BYTES);    // x chunk buffer 1

    const int tid  = threadIdx.x;
    const int lane = tid & 31;
    const int wv   = tid >> 5;
    const int nb   = blockIdx.x & 31;               // D2 / TILE_N = 32
    const int mb   = blockIdx.x >> 5;               // BATCH / TILE_M = 4
    const int nBase = nb * TILE_N;
    const int mBase = mb * TILE_M;

    // ---- One-time: convert + stage this block's W tile [TILE_N x D1] f32->f16 in LDS ----
    {
        const float* wg = W + (size_t)nBase * D1;
        #pragma unroll 1
        for (int g = 0; g < 16; ++g) {                 // 16 groups x 8 float4 per thread
            float4 f[8];
            #pragma unroll
            for (int i = 0; i < 8; ++i) {
                int idx = tid + (g * 8 + i) * 256;
                f[i] = *(const float4*)(wg + (size_t)(idx >> 9) * D1 + ((idx & 511) << 2));
            }
            #pragma unroll
            for (int i = 0; i < 8; ++i) {
                int idx = tid + (g * 8 + i) * 256;
                v4h h = { (_Float16)f[i].x, (_Float16)f[i].y, (_Float16)f[i].z, (_Float16)f[i].w };
                *(v4h*)(sW + (idx >> 9) * W_PITCH + ((idx & 511) << 2)) = h;
            }
        }
    }

    // ---- Wave -> (mSub pair, nSub) subtile assignment: 4x4 16x16 subtiles, 8 waves ----
    const int nSub  = wv & 3;
    const int mSub0 = (wv >> 2) * 2;
    const int mSub1 = mSub0 + 1;
    const int l15    = lane & 15;
    const int hiHalf = lane >> 4;        // 0 | 1
    const int aOff   = hiHalf * 8;       // A frag per-lane K offset (ISA 16-bit A layout)
    const int bOff   = hiHalf * 8;       // B frag per-lane K offset (ISA 16-bit B layout)

    const _Float16* bRowPtr   = sW  + (nSub  * 16 + l15) * W_PITCH;  // B col n = lane&15
    const _Float16* aRow0Ptr0 = sX0 + (mSub0 * 16 + l15) * X_PITCH;  // A row m = lane&15
    const _Float16* aRow1Ptr0 = sX0 + (mSub1 * 16 + l15) * X_PITCH;
    const _Float16* aRow0Ptr1 = sX1 + (mSub0 * 16 + l15) * X_PITCH;
    const _Float16* aRow1Ptr1 = sX1 + (mSub1 * 16 + l15) * X_PITCH;

    // Per-thread staging bases: column (tid&31)*4 is invariant; rows are tid>>5 + 8i.
    const int sRow = tid >> 5;
    const int sCol = (tid & 31) << 2;
    _Float16* ldst0 = sX0 + sRow * X_PITCH + sCol;
    _Float16* ldst1 = sX1 + sRow * X_PITCH + sCol;

    // Recurrent state lives in VGPRs for the entire T loop.
    v8f syn0 = {}, syn1 = {}, mem0 = {}, mem1 = {};

    const int nCol = nBase + nSub * 16 + l15;

    #pragma unroll 1
    for (int t = 0; t < T_STEPS; ++t) {
        const float* xgThread = x + ((size_t)t * BATCH + mBase + sRow) * D1 + sCol;
        v8f cur0 = {}, cur1 = {};
        XRegs r;

        // Software-pipelined, double-buffered K loop: one barrier per chunk.
        // Global clause for chunk k+1 is issued BEFORE the WMMAs of chunk k and
        // only consumed (cvt + ds_store) after them, so matrix math hides the
        // full global latency.
        stage_load(xgThread, r);
        stage_store(r, ldst0);
        __syncthreads();
        #pragma unroll 1
        for (int kc = 0; kc < NCHUNK; kc += 2) {
            if (kc + 1 < NCHUNK) stage_load(xgThread + (kc + 1) * KC, r);
            compute_chunk(bRowPtr, kc * KC, aRow0Ptr0, aRow1Ptr0, aOff, bOff, cur0, cur1);
            if (kc + 1 < NCHUNK) stage_store(r, ldst1);
            __syncthreads();
            if (kc + 2 < NCHUNK) stage_load(xgThread + (kc + 2) * KC, r);
            compute_chunk(bRowPtr, (kc + 1) * KC, aRow0Ptr1, aRow1Ptr1, aOff, bOff, cur0, cur1);
            if (kc + 2 < NCHUNK) stage_store(r, ldst0);
            __syncthreads();
        }

        // Synaptic LIF update (reset from *previous* mem, zero-reset) + spike store.
        // D-matrix layout: VGPR r -> row (r + 8*hiHalf), col = lane&15.
        #pragma unroll
        for (int rr = 0; rr < 8; ++rr) {
            {
                float reset = (mem0[rr] > 1.0f) ? 1.0f : 0.0f;
                float s = 0.8f * syn0[rr] + cur0[rr];
                float m = 0.9f * mem0[rr] + s;
                m -= reset * m;
                syn0[rr] = s; mem0[rr] = m;
                int bm = mBase + mSub0 * 16 + rr + hiHalf * 8;
                out[((size_t)t * BATCH + bm) * D2 + nCol] = (m > 1.0f) ? 1.0f : 0.0f;
            }
            {
                float reset = (mem1[rr] > 1.0f) ? 1.0f : 0.0f;
                float s = 0.8f * syn1[rr] + cur1[rr];
                float m = 0.9f * mem1[rr] + s;
                m -= reset * m;
                syn1[rr] = s; mem1[rr] = m;
                int bm = mBase + mSub1 * 16 + rr + hiHalf * 8;
                out[((size_t)t * BATCH + bm) * D2 + nCol] = (m > 1.0f) ? 1.0f : 0.0f;
            }
        }
    }
}

extern "C" void kernel_launch(void* const* d_in, const int* in_sizes, int n_in,
                              void* d_out, int out_size, void* d_ws, size_t ws_size,
                              hipStream_t stream) {
    (void)in_sizes; (void)n_in; (void)d_ws; (void)ws_size; (void)out_size;
    const float* x = (const float*)d_in[0];   // [T, B, D1] f32 (binary spikes)
    const float* W = (const float*)d_in[1];   // [D2, D1] f32
    float* out = (float*)d_out;               // [T, B, D2] f32 spikes

    hipFuncSetAttribute((const void*)snn_spiking_linear_kernel,
                        hipFuncAttributeMaxDynamicSharedMemorySize, SMEM_BYTES);

    dim3 grid((BATCH / TILE_M) * (D2 / TILE_N));   // 128 blocks
    dim3 block(256);                               // 8 wave32s
    snn_spiking_linear_kernel<<<grid, block, SMEM_BYTES, stream>>>(x, W, out);
}
